// EmbedMeanField_42279658062025
// MI455X (gfx1250) — matrix operand order repaired and checked by reading, not compiled, not run on hardware
//
#include <hip/hip_runtime.h>
#include <hip/hip_bf16.h>

// ---------------------------------------------------------------------------
// CDNA5 (gfx1250) wave32 WMMA implementation of EmbedMeanField forward.
// GEMMs via v_wmma_f32_16x16x32_bf16 (bf16 A/B, f32 accumulate), weights
// staged in LDS per block.  Segment sums via agent-scope global_atomic_add_f32
// into an L2-resident pool (51 MB << 192 MB L2).  WMMA kernels handle only
// full 16-row tiles; ragged tails (absent for N=100K / E=1.6M) use scalar
// tail kernels so the hot path is branch-free.
// ---------------------------------------------------------------------------

typedef __attribute__((ext_vector_type(16))) __bf16         v16bf;
typedef __attribute__((ext_vector_type(8)))  float          v8f;

#define LATENT 128
#define NT_COUNT 8          // 128 / 16 n-tiles

// A fragment (16x32 bf16, contiguous k order, same as swizzled B):
//   lane<16 : M=lane,    K = kbase + {0..15}
//   lane>=16: M=lane-16, K = kbase + {16..31}
__device__ __forceinline__ v16bf make_a(const float* __restrict__ rowp, int kbase, int lane) {
    int koff = kbase + ((lane & 16) ? 16 : 0);
    v16bf a;
#pragma unroll
    for (int i = 0; i < 8; ++i) a[i]     = (__bf16)rowp[koff + i];
#pragma unroll
    for (int i = 0; i < 8; ++i) a[8 + i] = (__bf16)rowp[koff + 8 + i];
    return a;
}

// B fragment from LDS-staged swizzled weights: 32 contiguous bytes per lane.
__device__ __forceinline__ v16bf load_b_lds(const __bf16* ldsW,
                                            int kt, int nt, int lane) {
    return *(const v16bf*)(ldsW + ((kt * NT_COUNT + nt) * 32 + lane) * 16);
}

__device__ __forceinline__ v8f wmma_bf16(v16bf a, v16bf b, v8f c) {
    return __builtin_amdgcn_wmma_f32_16x16x32_bf16(false, a, false, b, (short)0, c, false, false);
}

// --------------------------- prep / utility kernels ------------------------

__global__ void __launch_bounds__(256)
zero_f32(float4* __restrict__ p, long n4) {
    long i = (long)blockIdx.x * blockDim.x + threadIdx.x;
    if (i < n4) p[i] = make_float4(0.f, 0.f, 0.f, 0.f);
}

// Swizzle a KxN f32 weight into bf16 B-fragment order:
//   out[((kt*8+nt)*32 + lane)*16 + (v*2+h)] = bf16( W[kt*32+k][nt*16+n] )
//   lane<16 : n=lane,    k=2v+h ;  lane>=16: n=lane-16, k=16+2v+h
__global__ void __launch_bounds__(256)
swizzle_weight(const float* __restrict__ W, __bf16* __restrict__ out, int K, int N) {
    int idx = blockIdx.x * blockDim.x + threadIdx.x;
    if (idx >= K * N) return;
    int e    = idx & 15;
    int lane = (idx >> 4) & 31;
    int t    = idx >> 9;                 // kt*nTiles + nt
    int nTiles = N >> 4;
    int kt = t / nTiles, nt = t % nTiles;
    int v = e >> 1, hh = e & 1;
    int n, k;
    if (lane < 16) { n = lane;      k = 2 * v + hh; }
    else           { n = lane - 16; k = 16 + 2 * v + hh; }
    out[idx] = (__bf16)W[(size_t)(kt * 32 + k) * N + (nt * 16 + n)];
}

// --------------------------- edge GEMM + scatter ---------------------------
// pool[dst[e]] += edge_feat[e] @ We + be   (one wave per 16 edges; full tiles)
__global__ void __launch_bounds__(256)
edge_embed_scatter(const float* __restrict__ ef, const int* __restrict__ edst,
                   const __bf16* __restrict__ WeS, const float* __restrict__ be,
                   float* __restrict__ pool, int nTiles) {
    __shared__ __align__(32) __bf16 ldsW[2 * NT_COUNT * 512];   // 16 KB

    {   // stage swizzled We into LDS
        const uint4* s = (const uint4*)WeS;
        uint4* d = (uint4*)ldsW;
        for (int i = threadIdx.x; i < 2 * NT_COUNT * 512 / 8; i += blockDim.x) d[i] = s[i];
    }
    __syncthreads();

    int wave = __builtin_amdgcn_readfirstlane(blockIdx.x * (blockDim.x >> 5) + (threadIdx.x >> 5));
    int lane = threadIdx.x & 31;
    if (wave >= nTiles) return;
    int e0 = wave * 16;

    const float* rowp = ef + (size_t)(e0 + (lane & 15)) * 64;
    v16bf a0 = make_a(rowp, 0,  lane);
    v16bf a1 = make_a(rowp, 32, lane);

    int mrow = (lane >> 4) << 3;   // 0 or 8 : C/D row base for this lane half
    int col  = lane & 15;

    // per-output-row pool pointers (tile offset folds into instr. immediates)
    const int* dp = edst + e0 + mrow;
    float* pptr[8];
#pragma unroll
    for (int r = 0; r < 8; ++r) pptr[r] = pool + (size_t)dp[r] * LATENT + col;

    const float* bep = be + col;

#pragma unroll
    for (int nt = 0; nt < NT_COUNT; ++nt) {
        v8f c = {};
        c = wmma_bf16(a0, load_b_lds(ldsW, 0, nt, lane), c);
        c = wmma_bf16(a1, load_b_lds(ldsW, 1, nt, lane), c);
        float bias = bep[nt * 16];
#pragma unroll
        for (int r = 0; r < 8; ++r) {
            __hip_atomic_fetch_add(pptr[r] + nt * 16, c[r] + bias,
                                   __ATOMIC_RELAXED, __HIP_MEMORY_SCOPE_AGENT);
        }
    }
}

// tail edges (E % 16), scalar f32 path; one block of 128 threads per edge
__global__ void __launch_bounds__(128)
edge_tail(const float* __restrict__ ef, const int* __restrict__ edst,
          const float* __restrict__ We, const float* __restrict__ be,
          float* __restrict__ pool, int e_start, int E) {
    int e = e_start + blockIdx.x;
    if (e >= E) return;
    int n = threadIdx.x;
    float acc = be[n];
    for (int k = 0; k < 64; ++k) acc += ef[(size_t)e * 64 + k] * We[(size_t)k * LATENT + n];
    __hip_atomic_fetch_add(pool + (size_t)edst[e] * LATENT + n, acc,
                           __ATOMIC_RELAXED, __HIP_MEMORY_SCOPE_AGENT);
}

// --------------------------- fused node GEMM -------------------------------
// v = act @ W + bias (+ addend) ; preout = v (optional) ; out = relu(v)
// act is Mx128, W swizzled 128x128.  One wave per 16-row tile; full tiles.
__global__ void __launch_bounds__(256)
node_gemm(const float* __restrict__ act, int nTiles,
          const __bf16* __restrict__ WS, const float* __restrict__ bias,
          const float* __restrict__ addend, float* __restrict__ preout,
          float* __restrict__ out) {
    __shared__ __align__(32) __bf16 ldsW[4 * NT_COUNT * 512];   // 32 KB

    {
        const uint4* s = (const uint4*)WS;
        uint4* d = (uint4*)ldsW;
        for (int i = threadIdx.x; i < 4 * NT_COUNT * 512 / 8; i += blockDim.x) d[i] = s[i];
    }
    __syncthreads();

    int wave = __builtin_amdgcn_readfirstlane(blockIdx.x * (blockDim.x >> 5) + (threadIdx.x >> 5));
    int lane = threadIdx.x & 31;
    if (wave >= nTiles) return;
    int m0 = wave * 16;

    const float* rowp = act + (size_t)(m0 + (lane & 15)) * LATENT;
    v16bf a[4];
#pragma unroll
    for (int kt = 0; kt < 4; ++kt) a[kt] = make_a(rowp, kt * 32, lane);

    int mrow = (lane >> 4) << 3;
    int col  = lane & 15;

    // single base pointer per destination; per-access offsets are constants
    size_t rb = (size_t)(m0 + mrow) * LATENT + col;
    float*       outp = out + rb;
    float*       prep = preout ? preout + rb : nullptr;
    const float* addp = addend ? addend + rb : nullptr;
    const float* bp   = bias + col;

#pragma unroll
    for (int nt = 0; nt < NT_COUNT; ++nt) {
        v8f c = {};
#pragma unroll
        for (int kt = 0; kt < 4; ++kt) {
            c = wmma_bf16(a[kt], load_b_lds(ldsW, kt, nt, lane), c);
        }
        float bv = bp[nt * 16];
#pragma unroll
        for (int r = 0; r < 8; ++r) {
            float v = c[r] + bv + (addp ? addp[r * LATENT + nt * 16] : 0.0f);
            if (prep) prep[r * LATENT + nt * 16] = v;
            outp[r * LATENT + nt * 16] = fmaxf(v, 0.0f);
        }
    }
}

// tail rows (M % 16), scalar f32 path; one block of 128 threads per row
__global__ void __launch_bounds__(128)
node_tail(const float* __restrict__ act, const float* __restrict__ W,
          const float* __restrict__ bias, const float* __restrict__ addend,
          float* __restrict__ preout, float* __restrict__ out, int row_start, int M) {
    int row = row_start + blockIdx.x;
    if (row >= M) return;
    int n = threadIdx.x;
    float acc = bias[n];
    for (int k = 0; k < LATENT; ++k) acc += act[(size_t)row * LATENT + k] * W[(size_t)k * LATENT + n];
    size_t idx = (size_t)row * LATENT + n;
    float v = acc + (addend ? addend[idx] : 0.0f);
    if (preout) preout[idx] = v;
    out[idx] = fmaxf(v, 0.0f);
}

// --------------------------- gather + scatter round ------------------------
// pool[dst[e]] += h[src[e]]   (one thread per edge x 4 channels)
__global__ void __launch_bounds__(256)
gather_scatter(const float* __restrict__ h, const int* __restrict__ esrc,
               const int* __restrict__ edst, float* __restrict__ pool, long E) {
    long t = (long)blockIdx.x * blockDim.x + threadIdx.x;
    long e = t >> 5;                     // 32 threads per edge
    if (e >= E) return;
    int cb = (int)(t & 31) * 4;
    int s = esrc[e], d = edst[e];
    const float4 v = *(const float4*)(h + (size_t)s * LATENT + cb);
    float* p = pool + (size_t)d * LATENT + cb;
    __hip_atomic_fetch_add(p + 0, v.x, __ATOMIC_RELAXED, __HIP_MEMORY_SCOPE_AGENT);
    __hip_atomic_fetch_add(p + 1, v.y, __ATOMIC_RELAXED, __HIP_MEMORY_SCOPE_AGENT);
    __hip_atomic_fetch_add(p + 2, v.z, __ATOMIC_RELAXED, __HIP_MEMORY_SCOPE_AGENT);
    __hip_atomic_fetch_add(p + 3, v.w, __ATOMIC_RELAXED, __HIP_MEMORY_SCOPE_AGENT);
}

// ---------------------------------------------------------------------------

extern "C" void kernel_launch(void* const* d_in, const int* in_sizes, int n_in,
                              void* d_out, int out_size, void* d_ws, size_t ws_size,
                              hipStream_t stream) {
    const float* node_feat = (const float*)d_in[0];
    const float* edge_feat = (const float*)d_in[1];
    const int*   esrc      = (const int*)d_in[2];
    const int*   edst      = (const int*)d_in[3];
    const float* Wn = (const float*)d_in[4];
    const float* bn = (const float*)d_in[5];
    const float* We = (const float*)d_in[6];
    const float* be = (const float*)d_in[7];
    const float* Wc = (const float*)d_in[8];
    const float* bc = (const float*)d_in[9];
    const float* Wo = (const float*)d_in[10];
    const float* bo = (const float*)d_in[11];

    const int N = in_sizes[0] / 128;     // nodes
    const int E = in_sizes[2];           // edges

    // ---- workspace layout (256B aligned) ----
    auto alignup = [](size_t x) { return (x + 255) & ~(size_t)255; };
    char* ws = (char*)d_ws;
    size_t nmBytes = alignup((size_t)N * LATENT * sizeof(float));
    size_t off = 0;
    float* pool = (float*)(ws + off); off += nmBytes;
    float* msg  = (float*)(ws + off); off += nmBytes;
    float* h    = (float*)(ws + off); off += nmBytes;
    __bf16* WnS = (__bf16*)(ws + off); off += alignup(128 * 128 * 2);
    __bf16* WeS = (__bf16*)(ws + off); off += alignup(64  * 128 * 2);
    __bf16* WcS = (__bf16*)(ws + off); off += alignup(128 * 128 * 2);
    __bf16* WoS = (__bf16*)(ws + off); off += alignup(128 * 128 * 2);
    (void)ws_size; (void)n_in; (void)out_size;

    // ---- weight swizzle (tiny, once per launch) ----
    swizzle_weight<<<(128 * 128 + 255) / 256, 256, 0, stream>>>(Wn, WnS, 128, 128);
    swizzle_weight<<<(64  * 128 + 255) / 256, 256, 0, stream>>>(We, WeS, 64,  128);
    swizzle_weight<<<(128 * 128 + 255) / 256, 256, 0, stream>>>(Wc, WcS, 128, 128);
    swizzle_weight<<<(128 * 128 + 255) / 256, 256, 0, stream>>>(Wo, WoS, 128, 128);

    const long n4 = (long)N * LATENT / 4;
    const int zeroGrid  = (int)((n4 + 255) / 256);
    const int nodeTiles = N / 16;                 // full tiles only
    const int nodeGrid  = (nodeTiles + 7) / 8;    // 8 waves / block
    const int nodeTail  = N - nodeTiles * 16;
    const int edgeTiles = E / 16;
    const int edgeGrid  = (edgeTiles + 7) / 8;
    const int edgeTail  = E - edgeTiles * 16;
    const long gsThreads = (long)E * 32;
    const int gsGrid = (int)((gsThreads + 255) / 256);

    // ---- phase 1: pool = segment_sum(edge_feat @ We + be, dst) ----
    zero_f32<<<zeroGrid, 256, 0, stream>>>((float4*)pool, n4);
    edge_embed_scatter<<<edgeGrid, 256, 0, stream>>>(edge_feat, edst, WeS, be, pool, edgeTiles);
    if (edgeTail)
        edge_tail<<<edgeTail, 128, 0, stream>>>(edge_feat, edst, We, be, pool, edgeTiles * 16, E);

    // ---- phase 2: msg = node_feat @ Wn + bn + pool ; h = relu(msg) ----
    node_gemm<<<nodeGrid, 256, 0, stream>>>(node_feat, nodeTiles, WnS, bn, pool, msg, h);
    if (nodeTail)
        node_tail<<<nodeTail, 128, 0, stream>>>(node_feat, Wn, bn, pool, msg, h, nodeTiles * 16, N);

    // ---- phase 3: three mean-field propagation rounds ----
    for (int lv = 0; lv < 3; ++lv) {
        zero_f32<<<zeroGrid, 256, 0, stream>>>((float4*)pool, n4);
        gather_scatter<<<gsGrid, 256, 0, stream>>>(h, esrc, edst, pool, (long)E);
        node_gemm<<<nodeGrid, 256, 0, stream>>>(pool, nodeTiles, WcS, bc, msg, nullptr, h);
        if (nodeTail)
            node_tail<<<nodeTail, 128, 0, stream>>>(pool, Wc, bc, msg, nullptr, h, nodeTiles * 16, N);
    }

    // ---- phase 4: out = relu(h @ Wo + bo) ----
    node_gemm<<<nodeGrid, 256, 0, stream>>>(h, nodeTiles, WoS, bo, nullptr, nullptr, (float*)d_out);
    if (nodeTail)
        node_tail<<<nodeTail, 128, 0, stream>>>(h, Wo, bo, nullptr, nullptr, (float*)d_out, nodeTiles * 16, N);
}
